// DecoderLSTM_att_30580167147636
// MI455X (gfx1250) — compile-verified, matrix operand order checked by reading, not run on hardware
//
#include <hip/hip_runtime.h>
#include <hip/hip_bf16.h>

// ---- problem constants (match reference) ----
constexpr int kB  = 64;
constexpr int kN  = 512;
constexpr int kH  = 1024;
constexpr int kNH = 8;
constexpr int kHD = 128;      // kH / kNH
constexpr int kS  = 24;       // STEPS
constexpr int k3H = 3 * kH;   // 3072
constexpr int k4H = 4 * kH;   // 4096

typedef __attribute__((ext_vector_type(16))) __bf16 v16bf;
typedef __attribute__((ext_vector_type(8)))  __bf16 v8bf;
typedef __attribute__((ext_vector_type(8)))  float  v8f;

union ABFrag { v16bf v; v8bf h[2]; };

__device__ __forceinline__ float sigf(float x) { return 1.0f / (1.0f + __expf(-x)); }

// ---------------- conversion / init ----------------
__global__ void k_cvt_bf16(const float* __restrict__ src, __bf16* __restrict__ dst, int n) {
  int i = blockIdx.x * blockDim.x + threadIdx.x;
  if (i < n) dst[i] = (__bf16)src[i];
}

__global__ void k_init_state(const float* __restrict__ hidden, const float* __restrict__ cell,
                             const float* __restrict__ x0,
                             float* __restrict__ h, float* __restrict__ c,
                             __bf16* __restrict__ h_bf,
                             float* __restrict__ x, __bf16* __restrict__ x_bf) {
  int i = blockIdx.x * blockDim.x + threadIdx.x;
  if (i < kB * k3H) { float v = x0[i]; x[i] = v; x_bf[i] = (__bf16)v; }
  if (i < kB * kH)  { float hv = hidden[i]; h[i] = hv; h_bf[i] = (__bf16)hv; c[i] = cell[i]; }
}

// ---------------- bf16 WMMA GEMM: out[64,Nn] = A[64,K] @ W[Nn,K]^T + bias (+ out if beta) --
// Strip-mined: one wave owns a full-M 64x16 strip (4 accumulators). The weight fragment
// (the dominant, unique L2 stream) is loaded ONCE per k-chunk and reused by 4 WMMAs;
// A fragments are cache-resident (A is only 64 rows). A-frag lane layout per ISA:
// lane<16 holds row M=lane, K {0..7,16..23}; lane>=16 holds K {8..15,24..31}.
// W is [Nn,K] row-major = transpose-symmetric B layout (lane indexes the N column).
__global__ void k_gemm_bf16(const __bf16* __restrict__ A, const __bf16* __restrict__ W,
                            const float* __restrict__ bias, float* __restrict__ out,
                            int Nn, int K, int beta) {
  int wave = blockIdx.x * (blockDim.x >> 5) + (threadIdx.x >> 5);
  if (wave >= (Nn >> 4)) return;
  int lane = threadIdx.x & 31;
  int koff = (lane >> 4) << 3;
  int nr   = (wave << 4) + (lane & 15);
  const __bf16* ap = A + (size_t)(lane & 15) * K + koff;
  const __bf16* wp = W + (size_t)nr * K + koff;
  const size_t m16 = (size_t)16 * K;
  v8f acc[4] = {{}, {}, {}, {}};
  for (int k0 = 0; k0 < K; k0 += 32) {
    ABFrag b, a0, a1, a2, a3;
    b.h[0]  = *(const v8bf*)(wp + k0);
    b.h[1]  = *(const v8bf*)(wp + k0 + 16);
    a0.h[0] = *(const v8bf*)(ap + k0);
    a0.h[1] = *(const v8bf*)(ap + k0 + 16);
    a1.h[0] = *(const v8bf*)(ap + m16 + k0);
    a1.h[1] = *(const v8bf*)(ap + m16 + k0 + 16);
    a2.h[0] = *(const v8bf*)(ap + 2 * m16 + k0);
    a2.h[1] = *(const v8bf*)(ap + 2 * m16 + k0 + 16);
    a3.h[0] = *(const v8bf*)(ap + 3 * m16 + k0);
    a3.h[1] = *(const v8bf*)(ap + 3 * m16 + k0 + 16);
    acc[0] = __builtin_amdgcn_wmma_f32_16x16x32_bf16(false, a0.v, false, b.v, (short)0,
                                                     acc[0], false, false);
    acc[1] = __builtin_amdgcn_wmma_f32_16x16x32_bf16(false, a1.v, false, b.v, (short)0,
                                                     acc[1], false, false);
    acc[2] = __builtin_amdgcn_wmma_f32_16x16x32_bf16(false, a2.v, false, b.v, (short)0,
                                                     acc[2], false, false);
    acc[3] = __builtin_amdgcn_wmma_f32_16x16x32_bf16(false, a3.v, false, b.v, (short)0,
                                                     acc[3], false, false);
  }
  // D layout: VGPR r -> M = r (lanes 0-15) / r+8 (lanes 16-31); N = lane&15
  int rbase = (lane >> 4) << 3;
  int col   = (wave << 4) + (lane & 15);
  float bv  = bias ? bias[col] : 0.0f;
#pragma unroll
  for (int mt = 0; mt < 4; mt++) {
#pragma unroll
    for (int r = 0; r < 8; r++) {
      size_t o = (size_t)(mt * 16 + rbase + r) * Nn + col;
      float v = acc[mt][r] + bv;
      if (beta) v += out[o];
      out[o] = v;
    }
  }
}

// ---------------- LSTM pointwise (gate order i,f,g,o) ----------------
__global__ void k_lstm(const float* __restrict__ gates, float* __restrict__ h,
                       float* __restrict__ c, __bf16* __restrict__ h_bf) {
  int i = blockIdx.x * blockDim.x + threadIdx.x;
  if (i >= kB * kH) return;
  int b = i / kH, j = i % kH;
  const float* g = gates + (size_t)b * k4H;
  float ig = sigf(g[j]);
  float fg = sigf(g[j + kH]);
  float gg = tanhf(g[j + 2 * kH]);
  float og = sigf(g[j + 3 * kH]);
  float cn = fg * c[i] + ig * gg;
  float hn = og * tanhf(cn);
  c[i] = cn; h[i] = hn; h_bf[i] = (__bf16)hn;
}

// ---------------- split fused qkv projection into q + kv cache slot t ----------------
__global__ void k_qkv_scatter(const float* __restrict__ qkv, float* __restrict__ q,
                              float* __restrict__ kc, float* __restrict__ vc, int t) {
  int i = blockIdx.x * blockDim.x + threadIdx.x;
  if (i >= kB * kH) return;
  int b = i / kH, j = i % kH;
  const float* row = qkv + (size_t)b * k3H;
  q[i] = row[j];
  size_t o = ((size_t)b * kS + t) * kH + j;
  kc[o] = row[j + kH];
  vc[o] = row[j + 2 * kH];
}

// ---------------- causal attention over s<=t (tiny: fp32 VALU) ----------------
__global__ void k_attn(const float* __restrict__ q, const float* __restrict__ kc,
                       const float* __restrict__ vc, float* __restrict__ ctx,
                       __bf16* __restrict__ ctx_bf, int t) {
  __shared__ float red[kHD];
  __shared__ float sc[kS];
  int b = blockIdx.x / kNH, hh = blockIdx.x % kNH;
  int d = threadIdx.x;
  size_t qo = (size_t)b * kH + hh * kHD + d;
  float qv = q[qo];
  for (int s = 0; s <= t; s++) {
    red[d] = qv * kc[((size_t)b * kS + s) * kH + hh * kHD + d];
    __syncthreads();
    for (int off = kHD >> 1; off >= 1; off >>= 1) {
      if (d < off) red[d] += red[d + off];
      __syncthreads();
    }
    if (d == 0) sc[s] = red[0] * 0.08838834764831845f;  // 1/sqrt(128)
    __syncthreads();
  }
  float mx = -1e30f;
  for (int s = 0; s <= t; s++) mx = fmaxf(mx, sc[s]);
  float den = 0.0f;
  for (int s = 0; s <= t; s++) den += __expf(sc[s] - mx);
  float inv = 1.0f / den;
  float a = 0.0f;
  for (int s = 0; s <= t; s++) {
    float w = __expf(sc[s] - mx) * inv;
    a += w * vc[((size_t)b * kS + s) * kH + hh * kHD + d];
  }
  ctx[qo] = a;
  ctx_bf[qo] = (__bf16)a;
}

// ---------------- query input = 0.5*(h + attn_out) ----------------
__global__ void k_qin(const float* __restrict__ h, const float* __restrict__ attno,
                      __bf16* __restrict__ qin_bf) {
  int i = blockIdx.x * blockDim.x + threadIdx.x;
  if (i < kB * kH) qin_bf[i] = (__bf16)(0.5f * (h[i] + attno[i]));
}

// ---------------- pointer logits: batched GEMV, one wave per (b,n) ----------------
__global__ void k_logits(const float* __restrict__ query, const float* __restrict__ enc,
                         float* __restrict__ outl, int t) {
  int gw = (blockIdx.x * blockDim.x + threadIdx.x) >> 5;
  int lane = threadIdx.x & 31;
  if (gw >= kB * kN) return;
  int b = gw / kN, n = gw % kN;
  const float* qr = query + (size_t)b * kH;
  const float* er = enc + ((size_t)b * kN + n) * kH;
  float s = 0.0f;
  for (int j = lane; j < kH; j += 32) s += qr[j] * er[j];
#pragma unroll
  for (int off = 16; off >= 1; off >>= 1) s += __shfl_xor(s, off, 32);
  if (lane == 0) outl[((size_t)t * kB + b) * kN + n] = s;
}

// ---------------- top-3 (top_k semantics: desc value, lower index first), then sort asc ---
__global__ void k_top3(const float* __restrict__ outl, int* __restrict__ outidx,
                       int* __restrict__ idxbuf, int t) {
  int b = threadIdx.x;
  if (b >= kB) return;
  const float* row = outl + ((size_t)t * kB + b) * kN;
  int id0 = -1, id1 = -1, id2 = -1;
  for (int p = 0; p < 3; p++) {
    float best = -3.4e38f; int bi = -1;
    for (int n = 0; n < kN; n++) {
      if (n == id0 || n == id1) continue;
      float v = row[n];
      if (v > best) { best = v; bi = n; }
    }
    if (p == 0) id0 = bi; else if (p == 1) id1 = bi; else id2 = bi;
  }
  int a0 = id0, a1 = id1, a2 = id2, tmp;
  if (a0 > a1) { tmp = a0; a0 = a1; a1 = tmp; }
  if (a1 > a2) { tmp = a1; a1 = a2; a2 = tmp; }
  if (a0 > a1) { tmp = a0; a0 = a1; a1 = tmp; }
  size_t o = ((size_t)t * kB + b) * 3;
  outidx[o] = a0; outidx[o + 1] = a1; outidx[o + 2] = a2;
  idxbuf[b * 3] = a0; idxbuf[b * 3 + 1] = a1; idxbuf[b * 3 + 2] = a2;
}

// ---------------- gather next LSTM input x = enc[b, idx, :] flattened ----------------
__global__ void k_gather_x(const float* __restrict__ enc, const int* __restrict__ idxbuf,
                           float* __restrict__ x, __bf16* __restrict__ x_bf) {
  int i = blockIdx.x * blockDim.x + threadIdx.x;
  if (i >= kB * 3 * kH) return;
  int b = i / (3 * kH);
  int r = (i / kH) % 3;
  int j = i % kH;
  int id = idxbuf[b * 3 + r];
  float v = enc[((size_t)b * kN + id) * kH + j];
  x[i] = v; x_bf[i] = (__bf16)v;
}

extern "C" void kernel_launch(void* const* d_in, const int* in_sizes, int n_in,
                              void* d_out, int out_size, void* d_ws, size_t ws_size,
                              hipStream_t stream) {
  (void)in_sizes; (void)n_in; (void)out_size; (void)ws_size;
  const float* enc    = (const float*)d_in[0];
  const float* hidden = (const float*)d_in[1];
  const float* cell   = (const float*)d_in[2];
  /* d_in[3] end_node_embed: unused by reference */
  const float* x0     = (const float*)d_in[4];
  /* d_in[5] max_steps: compile-time kS */
  const float* W_ih = (const float*)d_in[6];
  const float* b_ih = (const float*)d_in[7];
  const float* W_hh = (const float*)d_in[8];
  const float* b_hh = (const float*)d_in[9];
  const float* Win  = (const float*)d_in[10];
  const float* bin  = (const float*)d_in[11];
  const float* Wout = (const float*)d_in[12];
  const float* bout = (const float*)d_in[13];
  const float* Wqt  = (const float*)d_in[14];
  const float* bqt  = (const float*)d_in[15];

  float* outl = (float*)d_out;                       // [S,B,N] logits
  int*   outi = (int*)(outl + (size_t)kS * kB * kN); // [S,B,3] indices

  // ---- workspace carve-out (all offsets 256B aligned) ----
  char* ws = (char*)d_ws;
  size_t off = 0;
  auto carve = [&](size_t bytes) -> char* {
    char* p = ws + off;
    off += (bytes + 255) & ~(size_t)255;
    return p;
  };
  __bf16* wih_bf  = (__bf16*)carve((size_t)k4H * k3H * 2);
  __bf16* whh_bf  = (__bf16*)carve((size_t)k4H * kH * 2);
  __bf16* win_bf  = (__bf16*)carve((size_t)k3H * kH * 2);
  __bf16* wout_bf = (__bf16*)carve((size_t)kH * kH * 2);
  __bf16* wqt_bf  = (__bf16*)carve((size_t)kH * kH * 2);
  float*  h       = (float*)carve((size_t)kB * kH * 4);
  float*  c       = (float*)carve((size_t)kB * kH * 4);
  __bf16* h_bf    = (__bf16*)carve((size_t)kB * kH * 2);
  float*  x       = (float*)carve((size_t)kB * k3H * 4);
  __bf16* x_bf    = (__bf16*)carve((size_t)kB * k3H * 2);
  float*  gates   = (float*)carve((size_t)kB * k4H * 4);
  float*  qkv     = (float*)carve((size_t)kB * k3H * 4);
  float*  q       = (float*)carve((size_t)kB * kH * 4);
  float*  kc      = (float*)carve((size_t)kB * kS * kH * 4);
  float*  vc      = (float*)carve((size_t)kB * kS * kH * 4);
  float*  ctx     = (float*)carve((size_t)kB * kH * 4);
  __bf16* ctx_bf  = (__bf16*)carve((size_t)kB * kH * 2);
  float*  attno   = (float*)carve((size_t)kB * kH * 4);
  __bf16* qin_bf  = (__bf16*)carve((size_t)kB * kH * 2);
  float*  query   = (float*)carve((size_t)kB * kH * 4);
  int*    idxbuf  = (int*)carve((size_t)kB * 3 * 4);

  auto cvt = [&](const float* s, __bf16* d, int n) {
    k_cvt_bf16<<<(n + 255) / 256, 256, 0, stream>>>(s, d, n);
  };
  auto gemm = [&](const __bf16* A, const __bf16* W, const float* bias, float* out,
                  int Nn, int K, int beta) {
    int waves = Nn / 16;  // one wave per 64x16 strip
    k_gemm_bf16<<<(waves + 7) / 8, 256, 0, stream>>>(A, W, bias, out, Nn, K, beta);
  };

  // ---- prologue: weights -> bf16 (L2-resident thereafter), init state ----
  cvt(W_ih, wih_bf, k4H * k3H);
  cvt(W_hh, whh_bf, k4H * kH);
  cvt(Win,  win_bf, k3H * kH);
  cvt(Wout, wout_bf, kH * kH);
  cvt(Wqt,  wqt_bf, kH * kH);
  k_init_state<<<(kB * k3H + 255) / 256, 256, 0, stream>>>(hidden, cell, x0, h, c, h_bf,
                                                           x, x_bf);

  // ---- 24 dependent decode steps; stream order provides inter-kernel sync ----
  for (int t = 0; t < kS; t++) {
    gemm(x_bf, wih_bf, b_ih, gates, k4H, k3H, 0);   // gates  = x @ W_ih^T + b_ih
    gemm(h_bf, whh_bf, b_hh, gates, k4H, kH, 1);    // gates += h @ W_hh^T + b_hh
    k_lstm<<<(kB * kH + 255) / 256, 256, 0, stream>>>(gates, h, c, h_bf);
    gemm(h_bf, win_bf, bin, qkv, k3H, kH, 0);       // fused q,k,v projection of h_t
    k_qkv_scatter<<<(kB * kH + 255) / 256, 256, 0, stream>>>(qkv, q, kc, vc, t);
    k_attn<<<kB * kNH, kHD, 0, stream>>>(q, kc, vc, ctx, ctx_bf, t);
    gemm(ctx_bf, wout_bf, bout, attno, kH, kH, 0);  // attention output proj
    k_qin<<<(kB * kH + 255) / 256, 256, 0, stream>>>(h, attno, qin_bf);
    gemm(qin_bf, wqt_bf, bqt, query, kH, kH, 0);    // query transform
    k_logits<<<(kB * kN) / 8, 256, 0, stream>>>(query, enc, outl, t);
    k_top3<<<1, 64, 0, stream>>>(outl, outi, idxbuf, t);
    k_gather_x<<<(kB * 3 * kH + 255) / 256, 256, 0, stream>>>(enc, idxbuf, x, x_bf);
  }
}
